// AttentionBlock_1065151889425
// MI455X (gfx1250) — compile-verified, matrix operand order checked by reading, not run on hardware
//
#include <hip/hip_runtime.h>
#include <math.h>

// ---------------------------------------------------------------------------
// Types for CDNA5 WMMA (wave32): A/B = 16 x bf16 per lane, C/D = 8 x f32.
// ---------------------------------------------------------------------------
typedef __bf16 bf16_t;
typedef __attribute__((ext_vector_type(8)))  bf16_t v8bf;
typedef __attribute__((ext_vector_type(16))) bf16_t v16bf;
typedef __attribute__((ext_vector_type(8)))  float  v8f;

static constexpr int Bn  = 4;     // batch
static constexpr int S   = 1024;  // sequence (32*32)
static constexpr int Cc  = 512;   // channels
static constexpr int NH  = 8;     // heads
static constexpr int DK  = 64;    // head dim
static constexpr int NQKV = 3 * NH * DK;  // 1536
static constexpr float kScale = 0.125f;   // 64^-0.5

// Fragment loader: memory is row-major [rows][ld] bf16. Produces the CDNA5
// 16-bit A/B fragment for a 16x32 (MxK) tile at (row0, k0):
//   lane L -> row (row0 + (L&15));  element j -> K = k0 + (j&7) + 16*(j>>3) + 8*(L>>4)
// i.e. two contiguous 16-byte loads per lane.
__device__ __forceinline__ v16bf load_frag(const bf16_t* __restrict__ base,
                                           int row0, int ld, int k0) {
  int lane = threadIdx.x & 31;
  int r = lane & 15, grp = lane >> 4;
  const bf16_t* p = base + (size_t)(row0 + r) * ld + k0 + grp * 8;
  v8bf lo = *(const v8bf*)p;
  v8bf hi = *(const v8bf*)(p + 16);
  union { v8bf h[2]; v16bf v; } u;
  u.h[0] = lo; u.h[1] = hi;
  return u.v;
}

__device__ __forceinline__ v8f wmma_bf16(v16bf a, v16bf b, v8f c) {
  // D(f32 16x16) = A(bf16 16x32) x B(bf16 32x16) + C
  return __builtin_amdgcn_wmma_f32_16x16x32_bf16(false, a, false, b,
                                                 (short)0, c, false, false);
}

__device__ __forceinline__ v8f vzero8f() {
  v8f z = {0.f, 0.f, 0.f, 0.f, 0.f, 0.f, 0.f, 0.f};
  return z;
}

// ---------------------------------------------------------------------------
// Prep kernels: fp32 -> bf16 weight convert; tiled transpose of x to xs[b][s][C]
// ---------------------------------------------------------------------------
__global__ void k_cvt(const float* __restrict__ in, bf16_t* __restrict__ out, int n) {
  int i = blockIdx.x * blockDim.x + threadIdx.x;
  if (i < n) out[i] = (bf16_t)in[i];
}

__global__ void k_txp(const float* __restrict__ x, bf16_t* __restrict__ xs) {
  __shared__ float tile[32][33];
  int b  = blockIdx.z;
  int i0 = blockIdx.x * 32;   // s
  int c0 = blockIdx.y * 32;   // C
  int tx = threadIdx.x, ty = threadIdx.y;
#pragma unroll
  for (int r = ty; r < 32; r += 8)
    tile[r][tx] = x[((size_t)b * Cc + c0 + r) * S + i0 + tx];
  __syncthreads();
#pragma unroll
  for (int r = ty; r < 32; r += 8)
    xs[((size_t)b * S + i0 + r) * Cc + c0 + tx] = (bf16_t)tile[tx][r];
}

// ---------------------------------------------------------------------------
// QKV projection: qkv[b, i, n] = xs[b,i,:] . Wp[n,:] + bp[n]
// One wave computes a 64x64 output tile (16 WMMA accumulators).
// Epilogue scatters into Q,K row-major [bh][s][64] and V transposed [bh][64][s].
// ---------------------------------------------------------------------------
__global__ __launch_bounds__(32) void k_qkv(const bf16_t* __restrict__ xs,
                                            const bf16_t* __restrict__ wp,
                                            const float*  __restrict__ bp,
                                            bf16_t* __restrict__ qk,
                                            bf16_t* __restrict__ vT) {
  int n0 = blockIdx.x * 64, m0 = blockIdx.y * 64, b = blockIdx.z;
  const bf16_t* A = xs + (size_t)b * S * Cc;

  v8f acc[4][4];
#pragma unroll
  for (int i = 0; i < 4; ++i)
#pragma unroll
    for (int j = 0; j < 4; ++j) acc[i][j] = vzero8f();

  for (int k0 = 0; k0 < Cc; k0 += 32) {
    v16bf af[4];
#pragma unroll
    for (int i = 0; i < 4; ++i) af[i] = load_frag(A, m0 + i * 16, Cc, k0);
#pragma unroll
    for (int j = 0; j < 4; ++j) {
      v16bf bfg = load_frag(wp, n0 + j * 16, Cc, k0);  // Wp rows are B^T rows
#pragma unroll
      for (int i = 0; i < 4; ++i) acc[i][j] = wmma_bf16(af[i], bfg, acc[i][j]);
    }
  }

  int lane = threadIdx.x & 31, col = lane & 15, grp = lane >> 4;
#pragma unroll
  for (int j = 0; j < 4; ++j) {
    int n = n0 + j * 16 + col;
    int h = n / 192, rr = n % 192;
    int kind = rr / 64, d = rr % 64;   // 0=Q, 1=K, 2=V
    int bh = b * NH + h;
    float bias = bp[n];
#pragma unroll
    for (int i = 0; i < 4; ++i) {
#pragma unroll
      for (int r = 0; r < 8; ++r) {
        int row = m0 + i * 16 + r + grp * 8;
        float v = acc[i][j][r] + bias;
        if (kind < 2)
          qk[(((size_t)bh * 2 + kind) * S + row) * DK + d] = (bf16_t)v;
        else
          vT[((size_t)bh * DK + d) * S + row] = (bf16_t)v;
      }
    }
  }
}

// ---------------------------------------------------------------------------
// Flash attention: each wave owns 16 query rows of one (b,h); online softmax.
// Row stats reduce across 16 lanes (C-fragment: row spans lanes of one VGPR).
// P converts C-layout -> A-layout via a 1KB per-wave LDS tile (in-order DS).
// ---------------------------------------------------------------------------
__global__ __launch_bounds__(128) void k_attn(const bf16_t* __restrict__ qk,
                                              const bf16_t* __restrict__ vT,
                                              bf16_t* __restrict__ res) {
  __shared__ __align__(16) bf16_t pbuf_all[4][16 * 32];
  int lane = threadIdx.x & 31, wave = threadIdx.x >> 5;
  int grp = lane >> 4, col = lane & 15;
  int bh = blockIdx.x;
  int b = bh >> 3, h = bh & 7;
  int q0 = (blockIdx.y * 4 + wave) * 16;

  const bf16_t* Q  = qk + ((size_t)bh * 2 + 0) * S * DK;
  const bf16_t* Km = qk + ((size_t)bh * 2 + 1) * S * DK;
  const bf16_t* Vt = vT + (size_t)bh * DK * S;
  bf16_t* pbuf = pbuf_all[wave];

  v16bf qa0 = load_frag(Q, q0, DK, 0);
  v16bf qa1 = load_frag(Q, q0, DK, 32);

  float m[8], l[8];
  v8f o[4];
#pragma unroll
  for (int r = 0; r < 8; ++r) { m[r] = -3.0e38f; l[r] = 0.f; }
#pragma unroll
  for (int d = 0; d < 4; ++d) o[d] = vzero8f();

  for (int j0 = 0; j0 < S; j0 += 32) {
    v8f s0 = vzero8f(), s1 = vzero8f();
    s0 = wmma_bf16(qa0, load_frag(Km, j0, DK, 0), s0);
    s0 = wmma_bf16(qa1, load_frag(Km, j0, DK, 32), s0);
    s1 = wmma_bf16(qa0, load_frag(Km, j0 + 16, DK, 0), s1);
    s1 = wmma_bf16(qa1, load_frag(Km, j0 + 16, DK, 32), s1);

#pragma unroll
    for (int r = 0; r < 8; ++r) {
      float a  = s0[r] * kScale;
      float c2 = s1[r] * kScale;
      float mx = fmaxf(a, c2);
#pragma unroll
      for (int sh = 1; sh < 16; sh <<= 1) mx = fmaxf(mx, __shfl_xor(mx, sh));
      float mn = fmaxf(m[r], mx);
      float pa = __expf(a - mn);
      float pb = __expf(c2 - mn);
      float sum = pa + pb;
#pragma unroll
      for (int sh = 1; sh < 16; sh <<= 1) sum += __shfl_xor(sum, sh);
      float alpha = __expf(m[r] - mn);
      m[r] = mn;
      l[r] = l[r] * alpha + sum;
#pragma unroll
      for (int d = 0; d < 4; ++d) o[d][r] = o[d][r] * alpha;
      int prow = r + grp * 8;
      pbuf[prow * 32 + col]      = (bf16_t)pa;
      pbuf[prow * 32 + 16 + col] = (bf16_t)pb;
    }

    v16bf pf = load_frag(pbuf, 0, 32, 0);  // P as A-matrix 16x32
#pragma unroll
    for (int d = 0; d < 4; ++d) {
      v16bf vb = load_frag(Vt + (size_t)(d * 16) * S, 0, S, j0);  // V^T rows = d
      o[d] = wmma_bf16(pf, vb, o[d]);
    }
  }

#pragma unroll
  for (int r = 0; r < 8; ++r) {
    float inv = 1.0f / l[r];
    int row = q0 + r + grp * 8;
#pragma unroll
    for (int d = 0; d < 4; ++d)
      res[((size_t)b * S + row) * Cc + h * DK + d * 16 + col] =
          (bf16_t)(o[d][r] * inv);
  }
}

// ---------------------------------------------------------------------------
// Output projection + bias + residual, written transposed to [B][C][s] fp32.
// ---------------------------------------------------------------------------
__global__ __launch_bounds__(32) void k_oproj(const bf16_t* __restrict__ res,
                                              const bf16_t* __restrict__ wo,
                                              const float*  __restrict__ bo,
                                              const float*  __restrict__ x,
                                              float* __restrict__ out) {
  int n0 = blockIdx.x * 64, m0 = blockIdx.y * 64, b = blockIdx.z;
  const bf16_t* A = res + (size_t)b * S * Cc;

  v8f acc[4][4];
#pragma unroll
  for (int i = 0; i < 4; ++i)
#pragma unroll
    for (int j = 0; j < 4; ++j) acc[i][j] = vzero8f();

  for (int k0 = 0; k0 < Cc; k0 += 32) {
    v16bf af[4];
#pragma unroll
    for (int i = 0; i < 4; ++i) af[i] = load_frag(A, m0 + i * 16, Cc, k0);
#pragma unroll
    for (int j = 0; j < 4; ++j) {
      v16bf bfg = load_frag(wo, n0 + j * 16, Cc, k0);  // Wo rows are B^T rows
#pragma unroll
      for (int i = 0; i < 4; ++i) acc[i][j] = wmma_bf16(af[i], bfg, acc[i][j]);
    }
  }

  int lane = threadIdx.x & 31, col = lane & 15, grp = lane >> 4;
#pragma unroll
  for (int j = 0; j < 4; ++j) {
    int c = n0 + j * 16 + col;
    float bias = bo[c];
#pragma unroll
    for (int i = 0; i < 4; ++i) {
#pragma unroll
      for (int r = 0; r < 8; ++r) {
        int row = m0 + i * 16 + r + grp * 8;
        size_t xi = ((size_t)b * Cc + c) * S + row;  // [B][C][s]
        out[xi] = acc[i][j][r] + bias + x[xi];       // residual = xs[b,row,c]
      }
    }
  }
}

// ---------------------------------------------------------------------------
extern "C" void kernel_launch(void* const* d_in, const int* in_sizes, int n_in,
                              void* d_out, int out_size, void* d_ws, size_t ws_size,
                              hipStream_t stream) {
  (void)in_sizes; (void)n_in; (void)out_size; (void)ws_size;
  const float* x  = (const float*)d_in[0];
  const float* Wp = (const float*)d_in[1];
  const float* bp = (const float*)d_in[2];
  const float* Wo = (const float*)d_in[3];
  const float* bo = (const float*)d_in[4];
  float* out = (float*)d_out;

  // Workspace layout (bf16): ~22 MB total, fits MI455X L2 with room to spare.
  bf16_t* ws  = (bf16_t*)d_ws;
  bf16_t* wpb = ws;  ws += (size_t)NQKV * Cc;            // 1536*512
  bf16_t* wob = ws;  ws += (size_t)Cc * Cc;              // 512*512
  bf16_t* xsb = ws;  ws += (size_t)Bn * S * Cc;          // 4*1024*512
  bf16_t* qk  = ws;  ws += (size_t)Bn * NH * 2 * S * DK; // Q,K row-major
  bf16_t* vT  = ws;  ws += (size_t)Bn * NH * DK * S;     // V transposed
  bf16_t* res = ws;                                      // 4*1024*512

  k_cvt<<<dim3((NQKV * Cc) / 256), 256, 0, stream>>>(Wp, wpb, NQKV * Cc);
  k_cvt<<<dim3((Cc * Cc) / 256), 256, 0, stream>>>(Wo, wob, Cc * Cc);
  k_txp<<<dim3(S / 32, Cc / 32, Bn), dim3(32, 8), 0, stream>>>(x, xsb);
  k_qkv<<<dim3(NQKV / 64, S / 64, Bn), 32, 0, stream>>>(xsb, wpb, bp, qk, vT);
  k_attn<<<dim3(Bn * NH, S / 64), 128, 0, stream>>>(qk, vT, res);
  k_oproj<<<dim3(Cc / 64, S / 64, Bn), 32, 0, stream>>>(res, wob, bo, x, out);
}
